// GAT_1168231104919
// MI455X (gfx1250) — matrix operand order repaired and checked by reading, not compiled, not run on hardware
//
#include <hip/hip_runtime.h>
#include <math.h>

#define F_IN    128
#define HEADS1  4
#define HID1    64
#define C1      (HEADS1 * HID1)   // 256
#define NCLASS  16
#define NEG_SLOPE 0.2f

typedef __attribute__((ext_vector_type(16))) _Float16 v16h;
typedef __attribute__((ext_vector_type(8)))  float    v8f;

// ---------------------------------------------------------------------------
// WMMA GEMM: C[M,N] = A[M,K] @ B[K,N]; f32 in/out, f16 multiply, f32 accumulate.
// One wave per 16x(16*NT) output strip. K and N are template constants so the
// K-loop fully unrolls and all strided B/C accesses fold into immediate offsets.
// A lane layout (contiguous, matching B): lanes 0-15 row=lane, K 0..15;
// lanes 16-31 row=lane-16, K 16..31.
// B lane layout (column-striped): lanes 0-15 col=lane, K 0..15; lanes 16-31 K 16..31.
// C/D layout: VGPR v -> row v (lanes 0-15) / v+8 (lanes 16-31), col = lane%16.
// ---------------------------------------------------------------------------
template <int K, int N, int NT>
__global__ void gat_wmma_gemm(const float* __restrict__ A, const float* __restrict__ B,
                              float* __restrict__ C) {
    const int lane = threadIdx.x & 31;
    const int m0   = blockIdx.x * 16;
    const int n0   = blockIdx.y * (16 * NT);
    const int arow = m0 + (lane & 15);
    const int kA   = (lane < 16) ? 0 : 16;
    const int lcol = lane & 15;
    const int kB   = (lane < 16) ? 0 : 16;

    v8f acc[NT];
#pragma unroll
    for (int t = 0; t < NT; ++t) acc[t] = (v8f){};

    const float* arow_base = A + (size_t)arow * K + kA;
    const float* bcol_base = B + (size_t)kB * N + n0 + lcol;

#pragma unroll
    for (int k0 = 0; k0 < K; k0 += 32) {
        // A fragment: one 64B-contiguous run per lane -> 4x float4 loads.
        const float* ap = arow_base + k0;
        const float4 a0 = *(const float4*)(ap);
        const float4 a1 = *(const float4*)(ap + 4);
        const float4 a2 = *(const float4*)(ap + 8);
        const float4 a3 = *(const float4*)(ap + 12);
        v16h a;
        a[0]  = (_Float16)a0.x; a[1]  = (_Float16)a0.y;
        a[2]  = (_Float16)a0.z; a[3]  = (_Float16)a0.w;
        a[4]  = (_Float16)a1.x; a[5]  = (_Float16)a1.y;
        a[6]  = (_Float16)a1.z; a[7]  = (_Float16)a1.w;
        a[8]  = (_Float16)a2.x; a[9]  = (_Float16)a2.y;
        a[10] = (_Float16)a2.z; a[11] = (_Float16)a2.w;
        a[12] = (_Float16)a3.x; a[13] = (_Float16)a3.y;
        a[14] = (_Float16)a3.z; a[15] = (_Float16)a3.w;

#pragma unroll
        for (int t = 0; t < NT; ++t) {
            // All 16 column loads share one base; k0, t, i are compile-time
            // constants -> immediate instruction offsets, no VALU address chain.
            const float* bp = bcol_base + (size_t)k0 * N + t * 16;
            v16h b;
#pragma unroll
            for (int i = 0; i < 16; ++i) b[i] = (_Float16)bp[i * N];
            acc[t] = __builtin_amdgcn_wmma_f32_16x16x32_f16(false, a, false, b,
                                                            (short)0, acc[t],
                                                            false, false);
        }
    }

    const int crow = m0 + ((lane < 16) ? 0 : 8);
    float* cp = C + (size_t)crow * N + n0 + lcol;
#pragma unroll
    for (int t = 0; t < NT; ++t)
#pragma unroll
        for (int v = 0; v < 8; ++v)
            cp[v * N + t * 16] = acc[t][v];
}

// ---------------------------------------------------------------------------
__global__ void gat_fill_f32(float* __restrict__ p, float v, int n) {
    int i = blockIdx.x * blockDim.x + threadIdx.x;
    if (i < n) p[i] = v;
}

// alpha_src[n,h] = sum_c h[n,h,c]*wsrc[h,c];  alpha_dst likewise. ch % 4 == 0.
__global__ void gat_alpha(const float* __restrict__ h, const float* __restrict__ wsrc,
                          const float* __restrict__ wdst, float* __restrict__ as,
                          float* __restrict__ ad, int n, int heads, int ch) {
    int tid = blockIdx.x * blockDim.x + threadIdx.x;
    if (tid >= n * heads) return;
    const int node = tid / heads, hh = tid - node * heads;
    const float* hp  = h + (size_t)node * heads * ch + (size_t)hh * ch;
    const float* ws_ = wsrc + hh * ch;
    const float* wd_ = wdst + hh * ch;
    float s = 0.f, d = 0.f;
    for (int c = 0; c < ch; c += 4) {
        float4 v  = *(const float4*)(hp + c);
        float4 w1 = *(const float4*)(ws_ + c);
        float4 w2 = *(const float4*)(wd_ + c);
        s += v.x * w1.x + v.y * w1.y + v.z * w1.z + v.w * w1.w;
        d += v.x * w2.x + v.y * w2.y + v.z * w2.z + v.w * w2.w;
    }
    as[tid] = s;
    ad[tid] = d;
}

__device__ __forceinline__ void gat_edge_sd(const int* __restrict__ esrc,
                                            const int* __restrict__ edst,
                                            int e, int Ereal, int& s, int& d) {
    if (e < Ereal) { s = esrc[e]; d = edst[e]; }
    else           { s = d = e - Ereal; }       // self-loop
}

__device__ __forceinline__ float gat_lrelu(float v) {
    return v > 0.f ? v : NEG_SLOPE * v;
}

// Pass A: per-destination max of leaky_relu(alpha_src[src]+alpha_dst[dst]).
__global__ void gat_edge_max(const int* __restrict__ esrc, const int* __restrict__ edst,
                             const float* __restrict__ as, const float* __restrict__ ad,
                             float* __restrict__ m, int etot, int Ereal, int heads) {
    int tid = blockIdx.x * blockDim.x + threadIdx.x;
    if (tid >= etot * heads) return;
    const int e = tid / heads, hh = tid - e * heads;
    int s, d;
    gat_edge_sd(esrc, edst, e, Ereal, s, d);
    float v = gat_lrelu(as[s * heads + hh] + ad[d * heads + hh]);
    atomicMax(&m[d * heads + hh], v);
}

// Pass B: ex = exp(e - m[dst]); per-destination sum; stash ex per edge.
__global__ void gat_edge_exp(const int* __restrict__ esrc, const int* __restrict__ edst,
                             const float* __restrict__ as, const float* __restrict__ ad,
                             const float* __restrict__ m, float* __restrict__ ssum,
                             float* __restrict__ ex, int etot, int Ereal, int heads) {
    int tid = blockIdx.x * blockDim.x + threadIdx.x;
    if (tid >= etot * heads) return;
    const int e = tid / heads, hh = tid - e * heads;
    int s, d;
    gat_edge_sd(esrc, edst, e, Ereal, s, d);
    float v  = gat_lrelu(as[s * heads + hh] + ad[d * heads + hh]);
    float xv = expf(v - m[d * heads + hh]);
    ex[tid] = xv;
    atomicAdd(&ssum[d * heads + hh], xv);
}

// Pass C: agg[dst, c] += h[src, c] * ex/(s+eps). One thread per (edge, 4-channel
// group): float4 gather of h[src], 4 coalesced f32 atomics into agg[dst].
__global__ void gat_edge_agg(const int* __restrict__ esrc, const int* __restrict__ edst,
                             const float* __restrict__ h, const float* __restrict__ ex,
                             const float* __restrict__ ssum, float* __restrict__ agg,
                             int etot, int Ereal, int heads, int ch) {
    const int C = heads * ch;
    const int G = C >> 2;                  // float4 groups per edge
    int tid = blockIdx.x * blockDim.x + threadIdx.x;
    if (tid >= etot * G) return;
    const int e = tid / G, g = tid - e * G;
    const int c = g << 2;
    const int hh = c / ch;                 // ch % 4 == 0 -> whole group same head
    int s, d;
    gat_edge_sd(esrc, edst, e, Ereal, s, d);
    float w = ex[(size_t)e * heads + hh] / (ssum[(size_t)d * heads + hh] + 1e-16f);
    const float4 hv = *(const float4*)(h + (size_t)s * C + c);
    float* ap = agg + (size_t)d * C + c;
    atomicAdd(ap + 0, hv.x * w);
    atomicAdd(ap + 1, hv.y * w);
    atomicAdd(ap + 2, hv.z * w);
    atomicAdd(ap + 3, hv.w * w);
}

// In-place bias + ELU on layer-1 aggregate.
__global__ void gat_bias_elu(float* __restrict__ a, const float* __restrict__ b,
                             int total, int C) {
    int tid = blockIdx.x * blockDim.x + threadIdx.x;
    if (tid >= total) return;
    float v = a[tid] + b[tid % C];
    a[tid] = v > 0.f ? v : (expf(v) - 1.f);
}

// Final: out = log_softmax(agg2 + b2) over NCLASS classes.
__global__ void gat_logsoftmax(const float* __restrict__ agg, const float* __restrict__ b,
                               float* __restrict__ out, int n) {
    int node = blockIdx.x * blockDim.x + threadIdx.x;
    if (node >= n) return;
    float v[NCLASS];
    float mx = -1e30f;
#pragma unroll
    for (int c = 0; c < NCLASS; ++c) {
        v[c] = agg[(size_t)node * NCLASS + c] + b[c];
        mx = fmaxf(mx, v[c]);
    }
    float sum = 0.f;
#pragma unroll
    for (int c = 0; c < NCLASS; ++c) sum += expf(v[c] - mx);
    float lse = mx + logf(sum);
#pragma unroll
    for (int c = 0; c < NCLASS; ++c)
        out[(size_t)node * NCLASS + c] = v[c] - lse;
}

// ---------------------------------------------------------------------------
extern "C" void kernel_launch(void* const* d_in, const int* in_sizes, int n_in,
                              void* d_out, int out_size, void* d_ws, size_t ws_size,
                              hipStream_t stream) {
    const float* x    = (const float*)d_in[0];
    const int*  edges = (const int*)  d_in[1];
    const float* W1   = (const float*)d_in[2];
    const float* as1w = (const float*)d_in[3];
    const float* ad1w = (const float*)d_in[4];
    const float* b1   = (const float*)d_in[5];
    const float* W2   = (const float*)d_in[6];
    const float* as2w = (const float*)d_in[7];
    const float* ad2w = (const float*)d_in[8];
    const float* b2   = (const float*)d_in[9];
    float* out = (float*)d_out;

    const int n    = in_sizes[0] / F_IN;   // 50000 (multiple of 16)
    const int E    = in_sizes[1] / 2;      // 800000
    const int etot = E + n;                // + self-loops
    const int* esrc = edges;
    const int* edst = edges + E;

    // Workspace layout (floats)
    float* ws = (float*)d_ws;
    size_t o = 0;
    float* h1   = ws + o; o += (size_t)n * C1;
    float* agg1 = ws + o; o += (size_t)n * C1;
    float* as1  = ws + o; o += (size_t)n * HEADS1;
    float* ad1  = ws + o; o += (size_t)n * HEADS1;
    float* m1   = ws + o; o += (size_t)n * HEADS1;
    float* s1   = ws + o; o += (size_t)n * HEADS1;
    float* ex1  = ws + o; o += (size_t)etot * HEADS1;
    float* h2   = ws + o; o += (size_t)n * NCLASS;
    float* as2  = ws + o; o += (size_t)n;
    float* ad2  = ws + o; o += (size_t)n;
    float* m2   = ws + o; o += (size_t)n;
    float* s2   = ws + o; o += (size_t)n;
    float* ex2  = ws + o; o += (size_t)etot;
    float* agg2 = ws + o; o += (size_t)n * NCLASS;

    const int TB = 256;
    auto blocks = [](long long work, int tb) { return (int)((work + tb - 1) / tb); };

    // ---------------- Layer 1 ----------------
    gat_wmma_gemm<F_IN, C1, 4><<<dim3(n / 16, C1 / 64), 32, 0, stream>>>(x, W1, h1);
    gat_alpha<<<blocks((long long)n * HEADS1, TB), TB, 0, stream>>>(
        h1, as1w, ad1w, as1, ad1, n, HEADS1, HID1);

    gat_fill_f32<<<blocks(n * HEADS1, TB), TB, 0, stream>>>(m1, -1e30f, n * HEADS1);
    gat_fill_f32<<<blocks(n * HEADS1, TB), TB, 0, stream>>>(s1, 0.f, n * HEADS1);
    gat_fill_f32<<<blocks((long long)n * C1, TB), TB, 0, stream>>>(agg1, 0.f, n * C1);

    gat_edge_max<<<blocks((long long)etot * HEADS1, TB), TB, 0, stream>>>(
        esrc, edst, as1, ad1, m1, etot, E, HEADS1);
    gat_edge_exp<<<blocks((long long)etot * HEADS1, TB), TB, 0, stream>>>(
        esrc, edst, as1, ad1, m1, s1, ex1, etot, E, HEADS1);
    gat_edge_agg<<<blocks((long long)etot * (C1 / 4), TB), TB, 0, stream>>>(
        esrc, edst, h1, ex1, s1, agg1, etot, E, HEADS1, HID1);

    gat_bias_elu<<<blocks((long long)n * C1, TB), TB, 0, stream>>>(agg1, b1, n * C1, C1);

    // ---------------- Layer 2 ----------------
    gat_wmma_gemm<C1, NCLASS, 1><<<dim3(n / 16, 1), 32, 0, stream>>>(agg1, W2, h2);
    gat_alpha<<<blocks(n, TB), TB, 0, stream>>>(h2, as2w, ad2w, as2, ad2, n, 1, NCLASS);

    gat_fill_f32<<<blocks(n, TB), TB, 0, stream>>>(m2, -1e30f, n);
    gat_fill_f32<<<blocks(n, TB), TB, 0, stream>>>(s2, 0.f, n);
    gat_fill_f32<<<blocks((long long)n * NCLASS, TB), TB, 0, stream>>>(agg2, 0.f, n * NCLASS);

    gat_edge_max<<<blocks(etot, TB), TB, 0, stream>>>(esrc, edst, as2, ad2, m2, etot, E, 1);
    gat_edge_exp<<<blocks(etot, TB), TB, 0, stream>>>(esrc, edst, as2, ad2, m2, s2, ex2,
                                                     etot, E, 1);
    gat_edge_agg<<<blocks((long long)etot * (NCLASS / 4), TB), TB, 0, stream>>>(
        esrc, edst, h2, ex2, s2, agg2, etot, E, 1, NCLASS);

    gat_logsoftmax<<<blocks(n, TB), TB, 0, stream>>>(agg2, b2, out, n);
}
